// PicoTrainableAttention_35631048687652
// MI455X (gfx1250) — matrix operand order repaired and checked by reading, not compiled, not run on hardware
//
#include <hip/hip_runtime.h>

// ---------------------------------------------------------------------------
// Causal MHA forward for MI455X (gfx1250, wave32, WMMA).
// B=2 S=2048 D=1024 H=16 HD=64.  d_out = [output f32 (4096x1024)] ++
// [attn_weights f32 (2*16*2048*2048)].  Workspace use: 48 MiB of d_ws.
//
// All matmuls: v_wmma_f32_16x16x32_bf16.  All WMMA operands are laid out so
// each lane's fragment bytes are contiguous (A row-major, B kept N-major /
// transposed) -> fragment loads lower to ds_load_b128 pairs, no scalar
// ds_load_u16 packing.  LDS staging uses the CDNA5 async-to-LDS path
// (global_load_async_to_lds_b128 + s_wait_asynccnt).
// ---------------------------------------------------------------------------

typedef __attribute__((ext_vector_type(16))) __bf16 v16bf;
typedef __attribute__((ext_vector_type(8)))  float  v8f;
typedef __attribute__((address_space(3)))    char   lds_char;

constexpr int kB  = 2;
constexpr int kS  = 2048;
constexpr int kD  = 1024;
constexpr int kH  = 16;
constexpr int kHD = 64;
constexpr int kM  = kB * kS;          // 4096 token rows
constexpr float kScale = 0.125f;      // 1/sqrt(64)

constexpr int TPITCH = 40;            // 32 + 8 pad (bf16 elems) for K=32 tiles

__device__ __forceinline__ v8f wmma_bf16(v16bf a, v16bf b, v8f c) {
  // D = A(16x32 bf16) * B(32x16 bf16) + C(16x16 f32)
  return __builtin_amdgcn_wmma_f32_16x16x32_bf16(
      /*neg_a=*/false, a, /*neg_b=*/false, b,
      /*c_mod=*/(short)0, c, /*reuse_a=*/false, /*reuse_b=*/false);
}

// Async 16-byte global -> LDS copy (per-lane), tracked by ASYNCcnt.
// CDNA5 ISA: GLOBAL_LOAD_ASYNC_TO_LDS_B128, GV mode.
__device__ __forceinline__ void async_copy16(const void* gsrc, void* ldst) {
  const unsigned lds_off = (unsigned)(__SIZE_TYPE__)(lds_char*)ldst;
  asm volatile("global_load_async_to_lds_b128 %0, %1, off"
               :: "v"(lds_off), "v"(gsrc) : "memory");
}
__device__ __forceinline__ void async_wait0() {
  asm volatile("s_wait_asynccnt 0x0" ::: "memory");
}

// A-matrix 16x32 bf16 fragment: lane L<16 holds row M=L, K=0..15;
// lane L+16 holds K=16..31 (same contiguous K order as the B fragment).
// 32 contiguous bytes per lane -> 2x ds_load_b128 / global_load_b128.
__device__ __forceinline__ v16bf load_a_frag(const __bf16* __restrict__ src, int ld) {
  const int lane = threadIdx.x & 31;
  const int m  = lane & 15;
  const int kb = (lane >> 4) << 4;     // 0 or 16
  const __bf16* p = src + (size_t)m * ld;
  v16bf a;
#pragma unroll
  for (int e = 0; e < 16; ++e) a[e] = p[kb + e];
  return a;
}

// Same A fragment, but source is f32 (attention probabilities) -> cvt to bf16.
__device__ __forceinline__ v16bf load_a_frag_f32(const float* __restrict__ src, int ld) {
  const int lane = threadIdx.x & 31;
  const int m  = lane & 15;
  const int kb = (lane >> 4) << 4;
  const float* p = src + (size_t)m * ld;
  v16bf a;
#pragma unroll
  for (int e = 0; e < 16; ++e) a[e] = (__bf16)p[kb + e];
  return a;
}

// B-matrix 32x16 fragment from an N-major (transposed) source:
// B(k,n) = src[n*ld + k].  Lane L<16 -> N=L, K=0..15; lane L+16 -> K=16..31.
// 32 contiguous bytes per lane -> 2x b128 loads.
__device__ __forceinline__ v16bf load_b_fragT(const __bf16* __restrict__ src, int ld) {
  const int lane = threadIdx.x & 31;
  const int n  = lane & 15;
  const int kb = (lane >> 4) << 4;     // 0 or 16
  const __bf16* p = src + (size_t)n * ld + kb;
  v16bf b;
#pragma unroll
  for (int e = 0; e < 16; ++e) b[e] = p[e];
  return b;
}

// ---------------------------------------------------------------------------
// f32 -> bf16 elementwise conversion (activations)
// ---------------------------------------------------------------------------
__global__ __launch_bounds__(256)
void cvt_f32_bf16_kernel(const float* __restrict__ in, __bf16* __restrict__ out, int n) {
  int i = blockIdx.x * 256 + threadIdx.x;
  if (i < n) out[i] = (__bf16)in[i];
}

// ---------------------------------------------------------------------------
// f32 [K][N] -> bf16 [N][K] tiled transpose-convert (weights, one-off).
// Block: 256 threads as 32x8; grid (K/32, N/32).
// ---------------------------------------------------------------------------
__global__ __launch_bounds__(256)
void cvt_transpose_kernel(const float* __restrict__ in, __bf16* __restrict__ out) {
  __shared__ float tile[32][33];
  const int k0 = blockIdx.x * 32;
  const int n0 = blockIdx.y * 32;
  const int tx = threadIdx.x & 31;
  const int ty = threadIdx.x >> 5;     // 0..7
#pragma unroll
  for (int i = 0; i < 32; i += 8)
    tile[ty + i][tx] = in[(size_t)(k0 + ty + i) * kD + n0 + tx];
  __syncthreads();
#pragma unroll
  for (int i = 0; i < 32; i += 8)
    out[(size_t)(n0 + ty + i) * kD + k0 + tx] = (__bf16)tile[tx][ty + i];
}

// ---------------------------------------------------------------------------
// Tiled WMMA GEMM: C(4096x1024) = A(4096x1024 bf16) * B, where B is supplied
// TRANSPOSED: BT[n][k] (1024x1024 bf16).
// mode 0: write bf16 into [B,H,S,HD] head-major layout (Q,K projections)
// mode 1: write f32 row-major (final output projection)
// mode 2: write bf16 into [B,H,HD,S] transposed layout (V projection)
// Tile: 128(M) x 64(N) per WG, BK=32, 8 waves as 4(M) x 2(N), each wave a
// 32x32 patch = 2x2 WMMA accumulators.
// ---------------------------------------------------------------------------
__global__ __launch_bounds__(256)
void gemm_bf16_kernel(const __bf16* __restrict__ A, const __bf16* __restrict__ BT,
                      float* __restrict__ Cf, __bf16* __restrict__ Cq, int mode) {
  __shared__ __attribute__((aligned(16))) __bf16 Asub [128 * TPITCH];  // [m][k]
  __shared__ __attribute__((aligned(16))) __bf16 BsubT[ 64 * TPITCH];  // [n][k]
  constexpr int K = kD, N = kD;
  const int m0 = blockIdx.x * 128;
  const int n0 = blockIdx.y * 64;
  const int t  = threadIdx.x;
  const int wave = t >> 5;
  const int wm = wave & 3;      // 0..3 along M
  const int wn = wave >> 2;     // 0..1 along N

  v8f acc00 = {}, acc01 = {}, acc10 = {}, acc11 = {};

  for (int kt = 0; kt < K / 32; ++kt) {
    const int k0 = kt * 32;
    __syncthreads();
    // Stage A 128x32 tile: 2 async 16B copies per thread.
#pragma unroll
    for (int i = 0; i < 2; ++i) {
      const int c   = t + i * 256;         // 0..511
      const int row = c >> 2;              // 0..127
      const int col = (c & 3) << 3;        // 0,8,16,24
      async_copy16(&A[(size_t)(m0 + row) * K + k0 + col],
                   &Asub[row * TPITCH + col]);
    }
    // Stage BT 64(n) x 32(k) tile: 1 async 16B copy per thread.
    {
      const int row = t >> 2;              // 0..63  (n)
      const int col = (t & 3) << 3;        // 0,8,16,24 (k)
      async_copy16(&BT[(size_t)(n0 + row) * K + k0 + col],
                   &BsubT[row * TPITCH + col]);
    }
    if (kt + 1 < K / 32)  // gfx1250 global_prefetch_b8 on next A tile
      __builtin_prefetch(&A[(size_t)(m0 + (t >> 1)) * K + k0 + 32], 0, 0);
    async_wait0();
    __syncthreads();

    v16bf a0 = load_a_frag (&Asub [(wm * 32 + 0)  * TPITCH], TPITCH);
    v16bf a1 = load_a_frag (&Asub [(wm * 32 + 16) * TPITCH], TPITCH);
    v16bf b0 = load_b_fragT(&BsubT[(wn * 32 + 0)  * TPITCH], TPITCH);
    v16bf b1 = load_b_fragT(&BsubT[(wn * 32 + 16) * TPITCH], TPITCH);
    acc00 = wmma_bf16(a0, b0, acc00);
    acc01 = wmma_bf16(a0, b1, acc01);
    acc10 = wmma_bf16(a1, b0, acc10);
    acc11 = wmma_bf16(a1, b1, acc11);
  }

  // C/D layout (ISA 7.12.2): N = lane&15, M = vgpr + 8*(lane>=16)
  const int lane = t & 31;
  const int nn = lane & 15;
  const int mb = (lane >> 4) << 3;
  v8f accs[4] = {acc00, acc01, acc10, acc11};
#pragma unroll
  for (int ti = 0; ti < 2; ++ti) {
#pragma unroll
    for (int tj = 0; tj < 2; ++tj) {
      v8f c = accs[ti * 2 + tj];
      const int gmb = m0 + wm * 32 + ti * 16 + mb;
      const int gn  = n0 + wn * 32 + tj * 16 + nn;
#pragma unroll
      for (int r = 0; r < 8; ++r) {
        const int gm = gmb + r;
        const int bb = gm >> 11, ss = gm & (kS - 1);
        const int hh = gn >> 6,  hd = gn & 63;
        if (mode == 0) {        // [B,H,S,HD] bf16 (Q, K)
          Cq[(((size_t)(bb * kH + hh) * kS + ss) << 6) + hd] = (__bf16)c[r];
        } else if (mode == 2) { // [B,H,HD,S] bf16 (V, transposed)
          Cq[((size_t)(bb * kH + hh) * kHD + hd) * kS + ss] = (__bf16)c[r];
        } else {                // f32 row-major
          Cf[(size_t)gm * N + gn] = c[r];
        }
      }
    }
  }
}

// ---------------------------------------------------------------------------
// scores[bh][q][k] = scale * dot(Q[bh][q], K[bh][k]) for k-tile <= q-tile.
// Grid: (B*H, S/128); 8 waves; wave w owns 16 q-rows. A-frags loaded once.
// K rows are [S,HD] row-major, so the B operand (K^T) is already N-major:
// per-lane reads are 32 contiguous bytes.
// ---------------------------------------------------------------------------
__global__ __launch_bounds__(256)
void attn_scores_kernel(const __bf16* __restrict__ Qall,
                        const __bf16* __restrict__ Kall,
                        float* __restrict__ attn) {
  const int bh   = blockIdx.x;
  const int qt   = blockIdx.y;
  const int wave = threadIdx.x >> 5;
  const int qr0  = qt * 128 + wave * 16;
  const __bf16* Qh = Qall + (size_t)bh * kS * kHD;
  const __bf16* Kh = Kall + (size_t)bh * kS * kHD;
  float* Ah = attn + (size_t)bh * kS * kS;

  const v16bf a0 = load_a_frag(Qh + (size_t)qr0 * kHD,      kHD);  // K-dim 0..31
  const v16bf a1 = load_a_frag(Qh + (size_t)qr0 * kHD + 32, kHD);  // K-dim 32..63

  const int ktmax = qr0 >> 4;           // causal: key tiles 0..qr0/16
  for (int kt = 0; kt <= ktmax; ++kt) {
    const int nt = kt * 16;
    if (kt < ktmax)
      __builtin_prefetch(Kh + (size_t)(nt + 16) * kHD, 0, 0);
    v16bf b0 = load_b_fragT(Kh + (size_t)nt * kHD,      kHD);
    v16bf b1 = load_b_fragT(Kh + (size_t)nt * kHD + 32, kHD);
    v8f acc = {};
    acc = wmma_bf16(a0, b0, acc);
    acc = wmma_bf16(a1, b1, acc);
    const int lane = threadIdx.x & 31;
    const int nn = lane & 15;
    const int mb = (lane >> 4) << 3;
    float* dst = Ah + (size_t)qr0 * kS + nt;
#pragma unroll
    for (int r = 0; r < 8; ++r)
      dst[(size_t)(mb + r) * kS + nn] = acc[r] * kScale;
  }
}

// ---------------------------------------------------------------------------
// In-place causal row softmax over attn rows of length S.  Row index =
// bh*S + q; valid keys k in [0, q]; k > q written as exact 0.
// ---------------------------------------------------------------------------
__global__ __launch_bounds__(256)
void softmax_rows_kernel(float* __restrict__ attn) {
  __shared__ float red[256];
  const int row = blockIdx.x;               // 0 .. B*H*S-1
  const int q   = row & (kS - 1);
  float* p = attn + (size_t)row * kS;
  const int n = q + 1;
  const int t = threadIdx.x;

  float m = -__builtin_inff();
  for (int k = t; k < n; k += 256) m = fmaxf(m, p[k]);
  red[t] = m; __syncthreads();
  for (int s2 = 128; s2 > 0; s2 >>= 1) {
    if (t < s2) red[t] = fmaxf(red[t], red[t + s2]);
    __syncthreads();
  }
  const float rmax = red[0];
  __syncthreads();

  float sum = 0.f;
  for (int k = t; k < n; k += 256) sum += __expf(p[k] - rmax);
  red[t] = sum; __syncthreads();
  for (int s2 = 128; s2 > 0; s2 >>= 1) {
    if (t < s2) red[t] += red[t + s2];
    __syncthreads();
  }
  const float rinv = 1.f / red[0];

  for (int k = t; k < kS; k += 256)
    p[k] = (k < n) ? __expf(p[k] - rmax) * rinv : 0.f;
}

// ---------------------------------------------------------------------------
// ctx[b][q][h*64+hd] = sum_k P[bh][q][k] * V[bh][k][hd].
// P is f32 (cvt to bf16 in the A-frag load).  V is stored TRANSPOSED
// [B,H,HD,S] so the staged LDS tile is N-major -> clean b128 frag loads.
// Masked region of P is exactly zero after softmax, so the full-range K loop
// is correct.  Grid: (B*H, S/128); wave w owns 16 q-rows.
// ---------------------------------------------------------------------------
__global__ __launch_bounds__(256)
void attn_pv_kernel(const float* __restrict__ attn,
                    const __bf16* __restrict__ Vt,   // [B,H,HD,S]
                    __bf16* __restrict__ ctx) {
  __shared__ __attribute__((aligned(16))) __bf16 VsubT[64 * TPITCH];  // [hd][k]
  const int bh = blockIdx.x;
  const int qt = blockIdx.y;
  const int bb = bh >> 4;        // H = 16
  const int hh = bh & 15;
  const int t  = threadIdx.x;
  const int wave = t >> 5;
  const int qr0  = qt * 128 + wave * 16;
  const float*  Ph  = attn + (size_t)bh * kS * kS;
  const __bf16* Vth = Vt   + (size_t)bh * kHD * kS;

  const v8f zero = {0.f, 0.f, 0.f, 0.f, 0.f, 0.f, 0.f, 0.f};
  v8f acc[4] = {zero, zero, zero, zero};

  const int ksteps = (qt + 1) * 4;  // keys [0, qt*128+128) in steps of 32
  for (int ks = 0; ks < ksteps; ++ks) {
    const int k0 = ks * 32;
    __syncthreads();
    {
      const int row = t >> 2;          // 0..63 (hd)
      const int col = (t & 3) << 3;    // 0,8,16,24 (k)
      async_copy16(&Vth[(size_t)row * kS + k0 + col],
                   &VsubT[row * TPITCH + col]);
    }
    async_wait0();
    __syncthreads();
    v16bf a  = load_a_frag_f32(Ph + (size_t)qr0 * kS + k0, kS);
    v16bf b0 = load_b_fragT(&VsubT[ 0 * TPITCH], TPITCH);
    v16bf b1 = load_b_fragT(&VsubT[16 * TPITCH], TPITCH);
    v16bf b2 = load_b_fragT(&VsubT[32 * TPITCH], TPITCH);
    v16bf b3 = load_b_fragT(&VsubT[48 * TPITCH], TPITCH);
    acc[0] = wmma_bf16(a, b0, acc[0]);
    acc[1] = wmma_bf16(a, b1, acc[1]);
    acc[2] = wmma_bf16(a, b2, acc[2]);
    acc[3] = wmma_bf16(a, b3, acc[3]);
  }

  const int lane = t & 31;
  const int nn = lane & 15;
  const int mb = (lane >> 4) << 3;
#pragma unroll
  for (int j = 0; j < 4; ++j) {
#pragma unroll
    for (int r = 0; r < 8; ++r) {
      const int q = qr0 + mb + r;
      ctx[(size_t)(bb * kS + q) * (kH * kHD) + hh * 64 + j * 16 + nn] =
          (__bf16)acc[j][r];
    }
  }
}

// ---------------------------------------------------------------------------
// Host-side orchestration.  Inputs: x, W_Q, W_K, W_V, W_O (all f32).
// Workspace layout: x_bf (8 MiB) | WqT/WkT/WvT/WoT bf16 (2 MiB each) |
// Q_bf | K_bf | Vt_bf | ctx_bf (8 MiB each).  Total 48 MiB.
// ---------------------------------------------------------------------------
extern "C" void kernel_launch(void* const* d_in, const int* in_sizes, int n_in,
                              void* d_out, int out_size, void* d_ws, size_t ws_size,
                              hipStream_t stream) {
  (void)in_sizes; (void)n_in; (void)out_size; (void)ws_size;
  const float* x  = (const float*)d_in[0];
  const float* Wq = (const float*)d_in[1];
  const float* Wk = (const float*)d_in[2];
  const float* Wv = (const float*)d_in[3];
  const float* Wo = (const float*)d_in[4];

  float* out  = (float*)d_out;                       // [B,S,D]
  float* attn = out + (size_t)kB * kS * kD;          // [B,H,S,S]

  char* w = (char*)d_ws;
  __bf16* x_bf   = (__bf16*)(w);
  __bf16* WqT_bf = (__bf16*)(w + (size_t)8 * 1024 * 1024);
  __bf16* WkT_bf = WqT_bf + (size_t)kD * kD;
  __bf16* WvT_bf = WkT_bf + (size_t)kD * kD;
  __bf16* WoT_bf = WvT_bf + (size_t)kD * kD;
  __bf16* Q_bf   = WoT_bf + (size_t)kD * kD;         // [B,H,S,HD]
  __bf16* K_bf   = Q_bf   + (size_t)kM * kD;         // [B,H,S,HD]
  __bf16* Vt_bf  = K_bf   + (size_t)kM * kD;         // [B,H,HD,S]
  __bf16* ctx_bf = Vt_bf  + (size_t)kM * kD;         // [B,S,H*HD]

  const int nX = kM * kD;        // 4,194,304
  cvt_f32_bf16_kernel<<<(nX + 255) / 256, 256, 0, stream>>>(x, x_bf, nX);
  const dim3 tr_grid(kD / 32, kD / 32);              // (32, 32)
  cvt_transpose_kernel<<<tr_grid, 256, 0, stream>>>(Wq, WqT_bf);
  cvt_transpose_kernel<<<tr_grid, 256, 0, stream>>>(Wk, WkT_bf);
  cvt_transpose_kernel<<<tr_grid, 256, 0, stream>>>(Wv, WvT_bf);
  cvt_transpose_kernel<<<tr_grid, 256, 0, stream>>>(Wo, WoT_bf);

  const dim3 gemm_grid(kM / 128, kD / 64);           // (32, 16)
  gemm_bf16_kernel<<<gemm_grid, 256, 0, stream>>>(x_bf, WqT_bf, nullptr, Q_bf, 0);
  gemm_bf16_kernel<<<gemm_grid, 256, 0, stream>>>(x_bf, WkT_bf, nullptr, K_bf, 0);
  gemm_bf16_kernel<<<gemm_grid, 256, 0, stream>>>(x_bf, WvT_bf, nullptr, Vt_bf, 2);

  const dim3 attn_grid(kB * kH, kS / 128);           // (32, 16)
  attn_scores_kernel<<<attn_grid, 256, 0, stream>>>(Q_bf, K_bf, attn);
  softmax_rows_kernel<<<kB * kH * kS, 256, 0, stream>>>(attn);
  attn_pv_kernel<<<attn_grid, 256, 0, stream>>>(attn, Vt_bf, ctx_bf);

  gemm_bf16_kernel<<<gemm_grid, 256, 0, stream>>>(ctx_bf, WoT_bf, out, nullptr, 1);
}
